// DiseaseAwareLSTMDecoder_11819749998887
// MI455X (gfx1250) — matrix operand order repaired and checked by reading, not compiled, not run on hardware
//
#include <hip/hip_runtime.h>
#include <hip/hip_bf16.h>
#include <cstdint>

// ---------------- shapes ----------------
#define BB   32      // batch
#define TT   60      // seq len
#define NPIX 49      // 7*7
#define CC   512     // feature channels / HID / E
#define DD   14      // disease dim
#define VV   10000   // vocab
#define VPAD 10048   // vocab padded to 64
#define GG   2048    // 4*HID
#define DCTX 256     // dis_ctx dim

typedef __attribute__((ext_vector_type(16))) __bf16 v16bf;
typedef __attribute__((ext_vector_type(8)))  __bf16 v8bf;
typedef __attribute__((ext_vector_type(8)))  float  v8f;

// ---- CDNA5 async global->LDS copy (16 bytes/lane), tracked by ASYNCcnt ----
__device__ __forceinline__ void async_copy_b128(const void* gptr, void* lptr)
{
    uint64_t ga = (uint64_t)(uintptr_t)gptr;
    uint32_t la = (uint32_t)(uintptr_t)lptr;   // low 32 bits = LDS byte address
    asm volatile("global_load_async_to_lds_b128 %0, %1, off"
                 :: "v"(la), "v"(ga) : "memory");
}
__device__ __forceinline__ void async_wait0()
{
    asm volatile("s_wait_asynccnt 0x0" ::: "memory");
}

// =====================================================================
// bf16 GEMM, f32 WMMA accumulate:
//   C[M,N] = A[M,K] @ B[K,Nb(cols 0..N-1 used)] (+ Cadd[M,N]) (+ biasN[N])
// Preconditions: M%32==0, K%64==0, Nb%64==0, 16B-aligned buffers.
// Block tile 32x64, BK=64, 8 waves -> one 16x16 tile each, 2 WMMA/iter.
// =====================================================================
__global__ __launch_bounds__(256)
void gemm_bf16_wmma(const __bf16* __restrict__ A, const __bf16* __restrict__ B,
                    float* __restrict__ C, int M, int N, int Nb, int K,
                    const float* __restrict__ Cadd, const float* __restrict__ biasN)
{
    __shared__ __align__(16) __bf16 sA[32 * 64];
    __shared__ __align__(16) __bf16 sB[64 * 64];

    const int tid  = threadIdx.x;
    const int wave = tid >> 5;
    const int lane = tid & 31;
    const int row0 = (wave >> 2) * 16;   // 0/16  : M within block
    const int col0 = (wave & 3) * 16;    // 0..48 : N within block
    const int bm   = blockIdx.y * 32;
    const int bn   = blockIdx.x * 64;

    const int arow = tid >> 3;           // 0..31
    const int acol = (tid & 7) * 8;      // 0..56
    const int brow = tid >> 2;           // 0..63
    const int bcol = (tid & 3) * 16;     // 0..48

    const int m    = lane & 15;
    const int half = lane >> 4;

    v8f acc = {};

    for (int k0 = 0; k0 < K; k0 += 64) {
        // ---- async fill of LDS tiles (3x b128 per lane) ----
        async_copy_b128(A + (size_t)(bm + arow) * K + k0 + acol,
                        sA + arow * 64 + acol);
        const __bf16* bp = B + (size_t)(k0 + brow) * Nb + bn + bcol;
        __bf16*       lp = sB + brow * 64 + bcol;
        async_copy_b128(bp,     lp);
        async_copy_b128(bp + 8, lp + 8);
        if (k0 + 64 < K)
            __builtin_prefetch(B + (size_t)(k0 + 64 + brow) * Nb + bn + bcol, 0, 1);
        async_wait0();
        __syncthreads();

        // ---- 2 x WMMA over the 64-deep K tile ----
        #pragma unroll
        for (int s = 0; s < 2; ++s) {
            v8bf alo = *(const v8bf*)(sA + (row0 + m) * 64 + s * 32 + half * 8);
            v8bf ahi = *(const v8bf*)(sA + (row0 + m) * 64 + s * 32 + half * 8 + 16);
            v16bf afrag;
            #pragma unroll
            for (int j = 0; j < 8; ++j) { afrag[j] = alo[j]; afrag[j + 8] = ahi[j]; }

            v16bf bfrag;
            #pragma unroll
            for (int j = 0; j < 8; ++j) {
                bfrag[j]     = sB[(s * 32 + half * 8 + j) * 64 + col0 + m];
                bfrag[j + 8] = sB[(s * 32 + half * 8 + 16 + j) * 64 + col0 + m];
            }

            acc = __builtin_amdgcn_wmma_f32_16x16x32_bf16(
                      false, afrag, false, bfrag, (short)0, acc, false, false);
        }
        __syncthreads();
    }

    // ---- epilogue: vgpr r -> M = r (+8 for upper half-wave), N = lane&15 ----
    const int gn = bn + col0 + (lane & 15);
    if (gn < N) {
        const int mbase = bm + row0 + half * 8;
        #pragma unroll
        for (int r = 0; r < 8; ++r) {
            int gm = mbase + r;
            float v = acc[r];
            if (Cadd)  v += Cadd[(size_t)gm * N + gn];
            if (biasN) v += biasN[gn];
            C[(size_t)gm * N + gn] = v;
        }
    }
}

// =====================================================================
// Disease MLPs: dis_att = relu(d@W_dp1+b)@W_dp2+b (32x512, ->bf16)
//               dis_ctx = relu(d@W_de1+b)@W_de2+b (32x256, ->bf16)
// =====================================================================
__global__ __launch_bounds__(256)
void mlp_kernel(const float* __restrict__ disease,
                const float* __restrict__ W_dp1, const float* __restrict__ b_dp1,
                const float* __restrict__ W_dp2, const float* __restrict__ b_dp2,
                const float* __restrict__ W_de1, const float* __restrict__ b_de1,
                const float* __restrict__ W_de2, const float* __restrict__ b_de2,
                __bf16* __restrict__ disattA, __bf16* __restrict__ disA)
{
    __shared__ float h1[128], h2[128];
    const int b = blockIdx.x, tid = threadIdx.x;
    if (tid < 128) {
        float a = b_dp1[tid], e = b_de1[tid];
        for (int d = 0; d < DD; ++d) {
            float x = disease[b * DD + d];
            a += x * W_dp1[d * 128 + tid];
            e += x * W_de1[d * 128 + tid];
        }
        h1[tid] = fmaxf(a, 0.f);
        h2[tid] = fmaxf(e, 0.f);
    }
    __syncthreads();
    for (int j = tid; j < CC; j += 256) {
        float a = b_dp2[j];
        for (int k = 0; k < 128; ++k) a += h1[k] * W_dp2[k * CC + j];
        disattA[b * CC + j] = (__bf16)a;
    }
    for (int j = tid; j < DCTX; j += 256) {
        float e = b_de2[j];
        for (int k = 0; k < 128; ++k) e += h2[k] * W_de2[k * DCTX + j];
        disA[b * DCTX + j] = (__bf16)e;
    }
}

// ---- f32 -> bf16 contiguous convert ----
__global__ void cvt_kernel(const float* __restrict__ src, __bf16* __restrict__ dst, int n)
{
    int i = blockIdx.x * blockDim.x + threadIdx.x;
    if (i < n) dst[i] = (__bf16)src[i];
}

// ---- f32 (K,N) -> bf16 (K,Npad), zero-padded columns ----
__global__ void cvtpad_kernel(const float* __restrict__ src, __bf16* __restrict__ dst,
                              int K, int N, int Npad)
{
    int i = blockIdx.x * blockDim.x + threadIdx.x;
    if (i >= K * Npad) return;
    int n = i % Npad, k = i / Npad;
    dst[i] = (n < N) ? (__bf16)src[(size_t)k * N + n] : (__bf16)0.f;
}

// ---- features (b,c,49) -> fbf rows b*49+n, cols c ----
__global__ void ftrans_kernel(const float* __restrict__ features, __bf16* __restrict__ fbf)
{
    int i = blockIdx.x * blockDim.x + threadIdx.x;      // BB*NPIX*CC
    if (i >= BB * NPIX * CC) return;
    int c = i & (CC - 1);
    int n = (i >> 9) % NPIX;
    int b = i / (NPIX * CC);
    fbf[i] = (__bf16)features[((size_t)b * CC + c) * NPIX + n];
}

// ---- emb[captions] -> bf16 rows t*32+b ----
__global__ void embgather_kernel(const int* __restrict__ captions,
                                 const float* __restrict__ emb, __bf16* __restrict__ embsA)
{
    int i = blockIdx.x * blockDim.x + threadIdx.x;      // TT*BB*CC
    if (i >= TT * BB * CC) return;
    int col = i & (CC - 1);
    int row = i >> 9;
    int t = row >> 5, b = row & 31;
    int tok = captions[b * TT + t];
    embsA[i] = (__bf16)emb[(size_t)tok * CC + col];
}

// ---- gbase[t*32+b][j] += disproj[b][j] + b_ih[j] + b_hh[j] (in place) ----
__global__ void gbase_kernel(float* __restrict__ gbase, const float* __restrict__ disproj,
                             const float* __restrict__ b_ih, const float* __restrict__ b_hh)
{
    int i = blockIdx.x * blockDim.x + threadIdx.x;      // TT*BB*GG
    if (i >= TT * BB * GG) return;
    int j = i & (GG - 1);
    int b = (i >> 11) & 31;
    gbase[i] += disproj[b * GG + j] + b_ih[j] + b_hh[j];
}

// ---- zero h, c, and h-half of xcat ----
__global__ void init_kernel(float* __restrict__ cstate, __bf16* __restrict__ hbuf,
                            __bf16* __restrict__ xcat)
{
    int i = blockIdx.x * blockDim.x + threadIdx.x;      // BB*CC
    if (i >= BB * CC) return;
    cstate[i] = 0.f;
    hbuf[i] = (__bf16)0.f;
    int b = i >> 9, j = i & (CC - 1);
    xcat[b * 1024 + CC + j] = (__bf16)0.f;
}

// =====================================================================
// Attention: scores = tanh(f_proj + attproj)@W_v + b_v ; softmax over 49;
// context = alpha @ f ; write bf16 context into xcat[:,0:512]
// =====================================================================
__global__ __launch_bounds__(256)
void attention_kernel(const float* __restrict__ f_proj, const float* __restrict__ attproj,
                      const float* __restrict__ W_v, const float* __restrict__ b_v,
                      const float* __restrict__ features, __bf16* __restrict__ xcat)
{
    __shared__ float sAp[CC];
    __shared__ float sScore[NPIX];
    __shared__ float sAlpha[NPIX];
    const int b = blockIdx.x, tid = threadIdx.x;
    for (int c = tid; c < CC; c += 256) sAp[c] = attproj[b * CC + c];
    __syncthreads();

    const int wave = tid >> 5, lane = tid & 31;
    for (int n = wave; n < NPIX; n += 8) {
        const float* fp = f_proj + (size_t)(b * NPIX + n) * CC;
        float s = 0.f;
        for (int c = lane; c < CC; c += 32) s += tanhf(fp[c] + sAp[c]) * W_v[c];
        for (int off = 16; off > 0; off >>= 1) s += __shfl_down(s, off, 32);
        if (lane == 0) sScore[n] = s + b_v[0];
    }
    __syncthreads();

    if (tid == 0) {
        float mx = -1e30f;
        for (int n = 0; n < NPIX; ++n) mx = fmaxf(mx, sScore[n]);
        float sum = 0.f;
        for (int n = 0; n < NPIX; ++n) { float e = __expf(sScore[n] - mx); sAlpha[n] = e; sum += e; }
        float inv = 1.f / sum;
        for (int n = 0; n < NPIX; ++n) sAlpha[n] *= inv;
    }
    __syncthreads();

    for (int c = tid; c < CC; c += 256) {
        const float* fb = features + ((size_t)b * CC + c) * NPIX;  // f[b,:,c] contiguous
        float ctx = 0.f;
        for (int n = 0; n < NPIX; ++n) ctx += sAlpha[n] * fb[n];
        xcat[b * 1024 + c] = (__bf16)ctx;
    }
}

// =====================================================================
// LSTM pointwise: gates (32x2048) -> c,h updates; h fanned out as bf16
// =====================================================================
__global__ __launch_bounds__(512)
void lstm_kernel(const float* __restrict__ gates, float* __restrict__ cstate,
                 __bf16* __restrict__ hbuf, __bf16* __restrict__ xcat,
                 __bf16* __restrict__ h_seq, int t)
{
    const int b = blockIdx.x, j = threadIdx.x;
    const float* g = gates + (size_t)b * GG;
    float gi = g[j], gf = g[CC + j], gg = g[2 * CC + j], go = g[3 * CC + j];
    float si = 1.f / (1.f + __expf(-gi));
    float sf = 1.f / (1.f + __expf(-gf));
    float so = 1.f / (1.f + __expf(-go));
    float cn = sf * cstate[b * CC + j] + si * tanhf(gg);
    float hn = so * tanhf(cn);
    cstate[b * CC + j] = cn;
    __bf16 hb = (__bf16)hn;
    hbuf[b * CC + j] = hb;
    xcat[b * 1024 + CC + j] = hb;
    h_seq[((size_t)b * TT + t) * CC + j] = hb;   // rows b*60+t for logits GEMM
}

// =====================================================================
extern "C" void kernel_launch(void* const* d_in, const int* in_sizes, int n_in,
                              void* d_out, int out_size, void* d_ws, size_t ws_size,
                              hipStream_t stream)
{
    const float* features = (const float*)d_in[0];
    const int*   captions = (const int*)  d_in[1];
    const float* disease  = (const float*)d_in[2];
    const float* emb      = (const float*)d_in[3];
    const float* W_attn   = (const float*)d_in[4];
    const float* b_attn   = (const float*)d_in[5];
    const float* W_v      = (const float*)d_in[6];
    const float* b_v      = (const float*)d_in[7];
    const float* W_dp1    = (const float*)d_in[8];
    const float* b_dp1    = (const float*)d_in[9];
    const float* W_dp2    = (const float*)d_in[10];
    const float* b_dp2    = (const float*)d_in[11];
    const float* W_de1    = (const float*)d_in[12];
    const float* b_de1    = (const float*)d_in[13];
    const float* W_de2    = (const float*)d_in[14];
    const float* b_de2    = (const float*)d_in[15];
    const float* W_ih     = (const float*)d_in[16];
    const float* W_hh     = (const float*)d_in[17];
    const float* b_ih     = (const float*)d_in[18];
    const float* b_hh     = (const float*)d_in[19];
    const float* W_fc     = (const float*)d_in[20];
    const float* b_fc     = (const float*)d_in[21];
    float* out = (float*)d_out;

    // ---- bump allocator over workspace ----
    char* wsp = (char*)d_ws;
    size_t off = 0;
    auto alloc = [&](size_t bytes) -> void* {
        void* p = wsp + off;
        off = (off + bytes + 255) & ~(size_t)255;
        return p;
    };
    __bf16* fbf     = (__bf16*)alloc((size_t)BB * NPIX * CC * 2);
    __bf16* Wfbf    = (__bf16*)alloc((size_t)CC * CC * 2);
    __bf16* Whbf    = (__bf16*)alloc((size_t)CC * CC * 2);
    __bf16* Wtop    = (__bf16*)alloc((size_t)CC * GG * 2);
    __bf16* Wcat    = (__bf16*)alloc((size_t)1024 * GG * 2);
    __bf16* Wdis    = (__bf16*)alloc((size_t)DCTX * GG * 2);
    __bf16* Wfcbf   = (__bf16*)alloc((size_t)CC * VPAD * 2);
    __bf16* embsA   = (__bf16*)alloc((size_t)TT * BB * CC * 2);
    __bf16* disA    = (__bf16*)alloc((size_t)BB * DCTX * 2);
    __bf16* disattA = (__bf16*)alloc((size_t)BB * CC * 2);
    float*  f_proj  = (float*)alloc((size_t)BB * NPIX * CC * 4);
    float*  gbase   = (float*)alloc((size_t)TT * BB * GG * 4);
    float*  disproj = (float*)alloc((size_t)BB * GG * 4);
    float*  att_base= (float*)alloc((size_t)BB * CC * 4);
    float*  attproj = (float*)alloc((size_t)BB * CC * 4);
    __bf16* xcat    = (__bf16*)alloc((size_t)BB * 1024 * 2);
    float*  gates   = (float*)alloc((size_t)BB * GG * 4);
    float*  cstate  = (float*)alloc((size_t)BB * CC * 4);
    __bf16* hbuf    = (__bf16*)alloc((size_t)BB * CC * 2);
    __bf16* h_seq   = (__bf16*)alloc((size_t)TT * BB * CC * 2);

    auto gemm = [&](const __bf16* A, const __bf16* B, float* C, int M, int N, int Nb, int K,
                    const float* Cadd, const float* biasN) {
        dim3 grid(Nb / 64, M / 32);
        gemm_bf16_wmma<<<grid, 256, 0, stream>>>(A, B, C, M, N, Nb, K, Cadd, biasN);
    };
    auto cvt = [&](const float* s, __bf16* d, int n) {
        cvt_kernel<<<(n + 255) / 256, 256, 0, stream>>>(s, d, n);
    };

    // ---- one-time weight conversions (all contiguous slices) ----
    cvt(W_attn,                 Wfbf,  CC * CC);          // W_attn[:512]
    cvt(W_attn + CC * CC,       Whbf,  CC * CC);          // W_attn[512:]
    cvt(W_ih,                   Wtop,  CC * GG);          // W_ih[0:512]
    cvt(W_ih + (size_t)CC * GG, Wcat,          CC * GG);  // W_ih[512:1024] -> Wcat top
    cvt(W_hh,                   Wcat + (size_t)CC * GG, CC * GG);  // W_hh -> Wcat bottom
    cvt(W_ih + (size_t)1024 * GG, Wdis, DCTX * GG);       // W_ih[1024:1280]
    {
        int n = CC * VPAD;
        cvtpad_kernel<<<(n + 255) / 256, 256, 0, stream>>>(W_fc, Wfcbf, CC, VV, VPAD);
    }

    int nft = BB * NPIX * CC;
    ftrans_kernel<<<(nft + 255) / 256, 256, 0, stream>>>(features, fbf);
    int nem = TT * BB * CC;
    embgather_kernel<<<(nem + 255) / 256, 256, 0, stream>>>(captions, emb, embsA);
    mlp_kernel<<<BB, 256, 0, stream>>>(disease, W_dp1, b_dp1, W_dp2, b_dp2,
                                       W_de1, b_de1, W_de2, b_de2, disattA, disA);
    init_kernel<<<(BB * CC + 255) / 256, 256, 0, stream>>>(cstate, hbuf, xcat);

    // ---- hoisted GEMMs ----
    gemm(fbf,     Wfbf, f_proj,  BB * NPIX, CC, CC, CC, nullptr, nullptr);   // f @ Wf
    gemm(embsA,   Wtop, gbase,   TT * BB,   GG, GG, CC, nullptr, nullptr);   // emb_t @ W_ih_top
    gemm(disA,    Wdis, disproj, BB,        GG, GG, DCTX, nullptr, nullptr); // dis_ctx @ W_ih_dis
    int ngb = TT * BB * GG;
    gbase_kernel<<<(ngb + 255) / 256, 256, 0, stream>>>(gbase, disproj, b_ih, b_hh);
    gemm(disattA, Whbf, att_base, BB, CC, CC, CC, nullptr, b_attn);          // dis_att @ Wh + b_attn

    // ---- recurrent loop: only h-dependent work remains ----
    for (int t = 0; t < TT; ++t) {
        gemm(hbuf, Whbf, attproj, BB, CC, CC, CC, att_base, nullptr);        // h@Wh + base
        attention_kernel<<<BB, 256, 0, stream>>>(f_proj, attproj, W_v, b_v, features, xcat);
        gemm(xcat, Wcat, gates, BB, GG, GG, 1024, gbase + (size_t)t * BB * GG, nullptr);
        lstm_kernel<<<BB, CC, 0, stream>>>(gates, cstate, hbuf, xcat, h_seq, t);
    }

    // ---- logits: h_seq (rows b*60+t) @ W_fc(padded) + b_fc -> out (32,60,10000) ----
    gemm(h_seq, Wfcbf, out, TT * BB, VV, VPAD, CC, nullptr, b_fc);
}
